// Ultimate_STCN_GraphSNN_63127429317305
// MI455X (gfx1250) — matrix operand order repaired and checked by reading, not compiled, not run on hardware
//
#include <hip/hip_runtime.h>

#define TT 240
#define NN 64
#define BB 512

typedef __attribute__((ext_vector_type(16))) _Float16 v16h;
typedef __attribute__((ext_vector_type(8)))  float    v8f;

// ---- dynamic LDS layout (float offsets) ----
#define OFF_XS    0        // 15360  X_seq[b] (240x64)
#define OFF_GX    15360    // 512    gx (64x8)
#define OFF_RED1  15872    // 128    red1t[16][8]  (w_s partials, transposed)
#define OFF_RED2  16000    // 128    red2t[8][16]  (w_skipgx partials, transposed)
#define OFF_SS    16128    // 16     spk_s
#define OFF_SHR   16144    // 48     spk_hr
#define OFF_XDEC  16192    // 80     DP-pool accumulator / decoder input
#define OFF_GDEC  16272    // 32     decoder hidden
#define OFF_CW    16304    // 120    conv_w
#define OFF_CB    16424    // 8      conv_b
#define OFF_L1S   16432    // 8      ln1_scale
#define OFF_L1B   16440    // 8      ln1_bias
#define OFF_FRR   16448    // 128    fr reduction
#define OFF_WS    16576    // 8192   w_sT      [16][512]
#define OFF_WGX   24768    // 4096   w_skipgxT [8][512]
#define OFF_WOUT  28864    // 384    w_outT    [8][48]
#define OFF_WSKS  29248    // 128    w_skipsT  [8][16]
#define OFF_END   29376    // 32-byte aligned (float offset % 8 == 0)
#define SMEM_BYTES (OFF_END * 4 + 2048)   // + 16x64 halves spkT (rows 8..15 zero pad)

// Issue one 16-byte async global->LDS transfer (tracked by ASYNCcnt).
__device__ __forceinline__ void async_b128(unsigned lds_byte, unsigned gl_off,
                                           const void* sbase) {
  asm volatile("global_load_async_to_lds_b128 %0, %1, %2"
               :: "v"(lds_byte), "v"(gl_off), "s"(sbase) : "memory");
}
__device__ __forceinline__ unsigned lds_off(const void* p) {
  return (unsigned)(unsigned long long)p;   // LDS aperture: offset = addr[31:0]
}

__global__ void __launch_bounds__(128)
snn_kernel(const float* __restrict__ Lg,       // (B,64,64)
           const float* __restrict__ Xg,       // (B,240,64)
           const float* __restrict__ convw,    // (15,1,1,8)
           const float* __restrict__ convb,    // (8)
           const float* __restrict__ ln1s, const float* __restrict__ ln1b,
           const float* __restrict__ w_s,      // (512,16)
           const float* __restrict__ w_skipgx, // (512,8)
           const float* __restrict__ w_skips,  // (16,8)
           const float* __restrict__ w_out,    // (48,8)
           const float* __restrict__ dw1,      // (80,32)
           const float* __restrict__ db1,      // (32)
           const float* __restrict__ dlns, const float* __restrict__ dlnb, // (32)
           const float* __restrict__ dw2,      // (32,4)
           const float* __restrict__ db2,      // (4)
           float* __restrict__ frpart,         // (B) per-block fr partials
           float* __restrict__ out)            // (B*4 + 1)
{
  extern __shared__ float sm[];
  float* Xs    = sm + OFF_XS;
  float* gx    = sm + OFF_GX;
  float* red1t = sm + OFF_RED1;
  float* red2t = sm + OFF_RED2;
  float* sS    = sm + OFF_SS;
  float* sHR   = sm + OFF_SHR;
  float* xdec  = sm + OFF_XDEC;
  float* gdec  = sm + OFF_GDEC;
  float* cw    = sm + OFF_CW;
  float* cb    = sm + OFF_CB;
  float* l1s   = sm + OFF_L1S;
  float* l1b   = sm + OFF_L1B;
  float* frr   = sm + OFF_FRR;
  float* wsT   = sm + OFF_WS;
  float* wgxT  = sm + OFF_WGX;
  float* woutT = sm + OFF_WOUT;
  float* wskT  = sm + OFF_WSKS;
  _Float16* spkT = (_Float16*)(sm + OFF_END);   // [16][64] f-major spikes, rows 8..15 = 0

  const int tid  = threadIdx.x;
  const int lane = tid & 31;
  const int wv   = tid >> 5;          // wave id 0..3 -> M-tile
  const int b    = blockIdx.x;
  const int hi   = (lane >= 16) ? 1 : 0;
  const int half = tid & 1;

  // ---- async-stage X_seq[b] (ASYNCcnt path); weights staged transposed ----
  const float* Xb = Xg + (size_t)b * (TT * NN);
  {
    const unsigned xs_b = lds_off(Xs);
    for (int i = tid; i < 3840; i += 128) async_b128(xs_b + i * 16, i * 16, Xb); // 61440 B
  }
  for (int i = tid; i < 8192; i += 128) wsT[(i & 15) * 512 + (i >> 4)] = w_s[i];
  for (int i = tid; i < 4096; i += 128) wgxT[(i & 7) * 512 + (i >> 3)] = w_skipgx[i];
  for (int i = tid; i < 384;  i += 128) woutT[(i & 7) * 48 + (i >> 3)] = w_out[i];
  if (tid < 128) wskT[(tid & 7) * 16 + (tid >> 3)] = w_skips[tid];
  for (int i = tid; i < 120;  i += 128) cw[i] = convw[i];
  if (tid < 8) { cb[tid] = convb[tid]; l1s[tid] = ln1s[tid]; l1b[tid] = ln1b[tid]; }
  if (tid < 80) xdec[tid] = 0.0f;
  for (int i = tid; i < 256; i += 128) ((unsigned*)spkT)[256 + i] = 0u; // pad rows 8..15

  // ---- preload this wave's A-tile of L_norm (rows wv*16..+15), f16, K-chunks 0/1 ----
  // 16-bit A layout: lane m=lane%16; VGPR v, pair p: K = (v<4 ? 2v : 2v+8) + p + hi*8
  const float* Lb = Lg + (size_t)b * (NN * NN);
  const int am = wv * 16 + (lane & 15);
  v16h a0, a1;
  #pragma unroll
  for (int e = 0; e < 16; ++e) {
    int v = e >> 1, p = e & 1;
    int kk = ((v < 4) ? 2 * v : 2 * v + 8) + p + hi * 8;
    a0[e] = (_Float16)Lb[am * NN + kk];
    a1[e] = (_Float16)Lb[am * NN + kk + 32];
  }

  // ---- per-thread LIF states ----
  float menc[4] = {0.f, 0.f, 0.f, 0.f};        // encoder mem: (n, fb..fb+3)
  float mem_s = 0.f, mh1 = 0.f, mh2 = 0.f, mh3 = 0.f, mout = 0.f; // tid<16 / tid<8
  float acc_se = 0.f, acc_ss = 0.f, acc_sh = 0.f, acc_so = 0.f;

  const int n  = tid >> 1;            // graph node 0..63
  const int fb = half * 4;            // channel base 0 or 4
  const int fcol = lane & 15;         // WMMA B/D column
  const _Float16* brow = spkT + fcol * 64 + hi * 8;   // this lane's B slice: K = hi*8+0..7, 16+hi*8+0..7

  asm volatile("s_wait_asynccnt 0x0" ::: "memory");   // async staging complete
  __syncthreads();

  // 15-tap sliding conv window in registers: xwin[k] = X[t + k - 7][n]
  float xwin[15];
  #pragma unroll
  for (int k = 0; k < 15; ++k) xwin[k] = (k >= 7) ? Xs[(k - 7) * NN + n] : 0.0f;

  const float4* cwp  = (const float4*)cw;   // cw[k*8 + fb .. +3] == cwp[k*2 + half]
  const float4  cb4  = ((const float4*)cb)[half];
  const float4  ls4  = ((const float4*)l1s)[half];
  const float4  lb4  = ((const float4*)l1b)[half];

  for (int t = 0; t < TT; ++t) {
    // ---- A: conv(15,SAME) + LayerNorm(8) + ReLU + encoder LIF ----
    float4 z4 = cb4;
    #pragma unroll
    for (int k = 0; k < 15; ++k) {
      float xv = xwin[k];
      float4 w = cwp[k * 2 + half];
      z4.x += xv * w.x; z4.y += xv * w.y; z4.z += xv * w.z; z4.w += xv * w.w;
    }
    float z[4]  = {z4.x, z4.y, z4.z, z4.w};
    float ls[4] = {ls4.x, ls4.y, ls4.z, ls4.w};
    float lb[4] = {lb4.x, lb4.y, lb4.z, lb4.w};
    float s4 = z[0] + z[1] + z[2] + z[3];
    float mu = (s4 + __shfl_xor(s4, 1)) * 0.125f;   // pair (tid^1) holds other 4 channels
    float q4 = (z[0]-mu)*(z[0]-mu) + (z[1]-mu)*(z[1]-mu)
             + (z[2]-mu)*(z[2]-mu) + (z[3]-mu)*(z[3]-mu);
    float inv = rsqrtf((q4 + __shfl_xor(q4, 1)) * 0.125f + 1e-5f);
    #pragma unroll
    for (int j = 0; j < 4; ++j) {
      float av = fmaxf((z[j] - mu) * inv * ls[j] + lb[j], 0.0f);
      float mv = fmaxf(menc[j] * 0.8f + av, -2.0f);
      float sp = (mv > 0.5f) ? 1.0f : 0.0f;
      menc[j] = mv - sp * 0.5f;
      acc_se += sp;
      spkT[(fb + j) * 64 + n] = (_Float16)sp;       // f-major, rows >=8 stay zero
    }
    // slide window (registers) + single LDS load for X[t+8]
    #pragma unroll
    for (int k = 0; k < 14; ++k) xwin[k] = xwin[k + 1];
    xwin[14] = (t + 8 < TT) ? Xs[(t + 8) * NN + n] : 0.0f;
    __syncthreads();                                   // bar1: spikes visible

    // ---- B: gx(64x8) = L(64x64) @ spk(64x8) via 2x v_wmma_f32_16x16x32_f16 ----
    typedef __attribute__((ext_vector_type(8))) _Float16 v8h;
    v16h b0v = __builtin_shufflevector(*(const v8h*)(brow), *(const v8h*)(brow + 16), 0,1,2,3,4,5,6,7,8,9,10,11,12,13,14,15);   // 2x 16B -> 2x ds_load_b128
    v16h b1v = __builtin_shufflevector(*(const v8h*)(brow + 32), *(const v8h*)(brow + 48), 0,1,2,3,4,5,6,7,8,9,10,11,12,13,14,15);
    v8f cacc = {};
    cacc = __builtin_amdgcn_wmma_f32_16x16x32_f16(false, a0, false, b0v, (short)0, cacc, false, false);
    cacc = __builtin_amdgcn_wmma_f32_16x16x32_f16(false, a1, false, b1v, (short)0, cacc, false, false);
    if (fcol < 8) {                       // D layout: VGPR r -> M = r + hi*8, N = lane%16
      int rbase = wv * 16 + hi * 8;
      #pragma unroll
      for (int r = 0; r < 8; ++r) gx[(rbase + r) * 8 + fcol] = cacc[r];
    }
    __syncthreads();                                   // bar2: gx visible

    // ---- C1: gx_flat@w_s (512->16) and gx_flat@w_skipgx (512->8) partials ----
    {
      int o1 = tid & 15, c1 = tid >> 4;               // 8 chunks x 64 rows
      const float4* gp = (const float4*)(gx + c1 * 64);
      const float4* wp = (const float4*)(wsT + o1 * 512 + c1 * 64);
      float pa = 0.f, pb = 0.f, pc = 0.f, pd = 0.f;
      #pragma unroll
      for (int j = 0; j < 16; ++j) {
        float4 g = gp[j], w = wp[j];
        pa += g.x * w.x; pb += g.y * w.y; pc += g.z * w.z; pd += g.w * w.w;
      }
      red1t[o1 * 8 + c1] = (pa + pb) + (pc + pd);
      int o2 = tid & 7, c2 = tid >> 3;                // 16 chunks x 32 rows
      const float4* gp2 = (const float4*)(gx + c2 * 32);
      const float4* wp2 = (const float4*)(wgxT + o2 * 512 + c2 * 32);
      float qa = 0.f, qb = 0.f, qc = 0.f, qd = 0.f;
      #pragma unroll
      for (int j = 0; j < 8; ++j) {
        float4 g = gp2[j], w = wp2[j];
        qa += g.x * w.x; qb += g.y * w.y; qc += g.z * w.z; qd += g.w * w.w;
      }
      red2t[o2 * 16 + c2] = (qa + qb) + (qc + qd);
    }
    __syncthreads();                                   // bar3: partials visible

    // ---- C2: spk_s LIF + hidden LIFs (wave0 lanes 0..15 own one unit each) ----
    if (tid < 16) {
      const float4* rp = (const float4*)(red1t + tid * 8);
      float4 r0 = rp[0], r1 = rp[1];
      float cs = ((r0.x + r0.y) + (r0.z + r0.w)) + ((r1.x + r1.y) + (r1.z + r1.w));
      mem_s = fmaxf(mem_s * 0.85f + cs, -2.0f);
      float ss = (mem_s > 0.3f) ? 1.0f : 0.0f;
      mem_s -= ss * 0.3f;
      sS[tid] = ss;  acc_ss += ss;
      mh1 = fmaxf(mh1 * 0.80f + ss, -2.0f); float s1 = (mh1 > 0.5f) ? 1.f : 0.f; mh1 -= s1 * 0.5f;
      mh2 = fmaxf(mh2 * 0.90f + ss, -2.0f); float s2 = (mh2 > 0.3f) ? 1.f : 0.f; mh2 -= s2 * 0.3f;
      mh3 = fmaxf(mh3 * 0.95f + ss, -2.0f); float s3 = (mh3 > 0.2f) ? 1.f : 0.f; mh3 -= s3 * 0.2f;
      sHR[tid] = s1; sHR[16 + tid] = s2; sHR[32 + tid] = s3;
      acc_sh += s1 + s2 + s3;
    }
    // C2 -> C3 is wave0-internal: same-wave LDS ops retire in order; just stop
    // the compiler from reordering (zero-cost scheduling fence).
    __builtin_amdgcn_wave_barrier();

    // ---- C3: output LIF + fused DP pooling (wave0 lanes 0..7, one channel each) ----
    if (tid < 8) {
      float ca = 0.f, cbb = 0.f;
      const float4* sh4 = (const float4*)sHR;
      const float4* wo4 = (const float4*)(woutT + tid * 48);
      #pragma unroll
      for (int j = 0; j < 12; ++j) {
        float4 s = sh4[j], w = wo4[j];
        ca += s.x * w.x + s.z * w.z; cbb += s.y * w.y + s.w * w.w;
      }
      const float4* ss4 = (const float4*)sS;
      const float4* wk4 = (const float4*)(wskT + tid * 16);
      #pragma unroll
      for (int j = 0; j < 4; ++j) {
        float4 s = ss4[j], w = wk4[j];
        ca += s.x * w.x + s.z * w.z; cbb += s.y * w.y + s.w * w.w;
      }
      const float4* r2 = (const float4*)(red2t + tid * 16);
      #pragma unroll
      for (int j = 0; j < 4; ++j) {
        float4 r = r2[j];
        ca += r.x + r.z; cbb += r.y + r.w;
      }
      float co = ca + cbb;
      mout = fmaxf(mout * 0.9f + co, -2.0f);
      float so = (mout > 0.3f) ? 1.0f : 0.0f;
      mout -= so * 0.3f;
      acc_so += so;
      int ch = t / 24, pp = t % 24;     // -mean(first 12) + mean(last 12)
      float coef = (pp >= 12) ? (1.0f / 12.0f) : (-1.0f / 12.0f);
      xdec[ch * 8 + tid] += coef * mout;
    }
    // no loop-end barrier needed: all cross-wave hazards covered by bar1..bar3
  }

  // ---- decoder: (80)->dec_w1->LN->GELU->(32)->dec_w2->(4) ----
  if (tid < 32) {                        // wave 0 only (xdec written by wave 0, in order)
    float y = db1[tid];
    #pragma unroll 8
    for (int i = 0; i < 80; ++i) y += xdec[i] * dw1[i * 32 + tid];
    float s = y;
    #pragma unroll
    for (int mk = 16; mk >= 1; mk >>= 1) s += __shfl_xor(s, mk);
    float mu = s * (1.0f / 32.0f);
    float d = y - mu, q = d * d;
    #pragma unroll
    for (int mk = 16; mk >= 1; mk >>= 1) q += __shfl_xor(q, mk);
    float inv = rsqrtf(q * (1.0f / 32.0f) + 1e-5f);
    float zz = d * inv * dlns[tid] + dlnb[tid];
    float g = 0.5f * zz * (1.0f + tanhf(0.7978845608028654f * (zz + 0.044715f * zz * zz * zz)));
    gdec[tid] = g;
  }
  __syncthreads();
  if (tid < 4) {
    float o = db2[tid];
    #pragma unroll
    for (int i = 0; i < 32; ++i) o += gdec[i] * dw2[i * 4 + tid];
    out[b * 4 + tid] = o;
  }

  // ---- firing-rate partial for this block (deterministic 2-pass) ----
  frr[tid] = acc_se * (1.0f / 512.0f) + acc_ss * (1.0f / 16.0f)
           + acc_sh * (1.0f / 48.0f) + acc_so * (1.0f / 8.0f);
  __syncthreads();
  #pragma unroll
  for (int s = 64; s > 0; s >>= 1) {
    if (tid < s) frr[tid] += frr[tid + s];
    __syncthreads();
  }
  if (tid == 0) frpart[b] = frr[0];
}

__global__ void __launch_bounds__(256)
fr_reduce_kernel(const float* __restrict__ frpart, float* __restrict__ out) {
  __shared__ float r[256];
  int tid = threadIdx.x;
  r[tid] = frpart[tid] + frpart[tid + 256];
  __syncthreads();
  for (int s = 128; s > 0; s >>= 1) {
    if (tid < s) r[tid] += r[tid + s];
    __syncthreads();
  }
  if (tid == 0) out[BB * 4] = r[0] * (1.0f / (4.0f * (float)TT * (float)BB));
}

extern "C" void kernel_launch(void* const* d_in, const int* in_sizes, int n_in,
                              void* d_out, int out_size, void* d_ws, size_t ws_size,
                              hipStream_t stream) {
  const float* Lg       = (const float*)d_in[0];
  const float* Xg       = (const float*)d_in[1];
  const float* convw    = (const float*)d_in[2];
  const float* convb    = (const float*)d_in[3];
  const float* ln1s     = (const float*)d_in[4];
  const float* ln1b     = (const float*)d_in[5];
  const float* w_s      = (const float*)d_in[6];
  const float* w_skipgx = (const float*)d_in[7];
  const float* w_skips  = (const float*)d_in[8];
  const float* w_out    = (const float*)d_in[9];
  const float* dw1      = (const float*)d_in[10];
  const float* db1      = (const float*)d_in[11];
  const float* dlns     = (const float*)d_in[12];
  const float* dlnb     = (const float*)d_in[13];
  const float* dw2      = (const float*)d_in[14];
  const float* db2      = (const float*)d_in[15];
  float* out = (float*)d_out;
  float* frp = (float*)d_ws;   // BB floats of scratch

  snn_kernel<<<dim3(BB), dim3(128), SMEM_BYTES, stream>>>(
      Lg, Xg, convw, convb, ln1s, ln1b, w_s, w_skipgx, w_skips, w_out,
      dw1, db1, dlns, dlnb, dw2, db2, frp, out);
  fr_reduce_kernel<<<dim3(1), dim3(256), 0, stream>>>(frp, out);
}
